// Model_52329881534787
// MI455X (gfx1250) — matrix operand order repaired and checked by reading, not compiled, not run on hardware
//
#include <hip/hip_runtime.h>
#include <hip/hip_bf16.h>

// ---------------- problem constants ----------------
#define H      1024
#define G4     4096          // 4*H
#define OUTV   32000
#define SEQ    64
#define DSTEPS 12            // MAX_LEN-1

typedef __attribute__((ext_vector_type(16))) _Float16 v16h;
typedef __attribute__((ext_vector_type(8)))  _Float16 v8h;
typedef __attribute__((ext_vector_type(8)))  float    v8f;

// ---------------- workspace layout (bytes) ----------------
// f16 weight copies (L2-resident working set ~96MB) + small state
#define WS_ENC_WIH   0ull                               // 4096*1024*2 = 8 MiB
#define WS_ENC_WHH   (WS_ENC_WIH + 8388608ull)
#define WS_DEC_WIH   (WS_ENC_WHH + 8388608ull)
#define WS_DEC_WHH   (WS_DEC_WIH + 8388608ull)
#define WS_OUTW      (WS_DEC_WHH + 8388608ull)          // 32000*1024*2
#define WS_GATES     (WS_OUTW + 65536000ull)            // 4096 f32
#define WS_C         (WS_GATES + 16384ull)              // 1024 f32
#define WS_H16       (WS_C + 4096ull)                   // 1024 f16
#define WS_LOGITS    (WS_H16 + 2048ull)                 // 32000 f32
#define WS_TOK       (WS_LOGITS + 128000ull)            // int
#define WS_DONE      (WS_TOK + 4ull)                    // int

// ---------------- f32 -> f16 weight conversion ----------------
__global__ void cvt_f16_kernel(const float* __restrict__ src,
                               _Float16* __restrict__ dst, int n) {
    int i = blockIdx.x * blockDim.x + threadIdx.x;
    int stride = gridDim.x * blockDim.x;
    for (; i < n; i += stride) dst[i] = (_Float16)src[i];
}

// ---------------- state init ----------------
__global__ void init_kernel(_Float16* h16, float* c, int* tok, int* done,
                            float* dout) {
    int i = threadIdx.x;           // 1024 threads
    h16[i] = (_Float16)0.f;
    c[i]   = 0.f;
    if (i == 0) { *tok = 0; *done = 0; dout[0] = 0.f; /* SOS token */ }
}

// ---------------- LSTM gates via WMMA ----------------
// gates[4096] = Wih@x + b_ih + Whh@h + b_hh,  x = (relu?)(emb[*tok_addr])
// grid = 32 blocks x 256 threads (8 waves); each wave owns 16 gate rows.
__global__ __launch_bounds__(256)
void gates_wmma_kernel(const _Float16* __restrict__ Wih,
                       const _Float16* __restrict__ Whh,
                       const float* __restrict__ b_ih,
                       const float* __restrict__ b_hh,
                       const float* __restrict__ emb,
                       const int* __restrict__ tok_addr,
                       const _Float16* __restrict__ h16,
                       float* __restrict__ gates,
                       int apply_relu) {
    __shared__ __align__(32) _Float16 xs[H];
    __shared__ __align__(32) _Float16 hs[H];
    const int tid = threadIdx.x;
    const int tok = *tok_addr;
    const float* __restrict__ erow = emb + (size_t)tok * H;
    for (int i = tid; i < H; i += 256) {
        float v = erow[i];
        if (apply_relu) v = v > 0.f ? v : 0.f;
        xs[i] = (_Float16)v;
        hs[i] = h16[i];
    }
    __syncthreads();

    const int wave = tid >> 5, lane = tid & 31;
    const int j0    = (blockIdx.x * 8 + wave) * 16;          // 16 gate rows
    const int rowA  = j0 + (lane & 15);
    const int koffA = (lane >> 4) * 16;                      // A frag K-offset
    const int koffB = (lane >> 4) * 16;                      // B frag K-offset
    const _Float16* __restrict__ wi = Wih + (size_t)rowA * H + koffA;
    const _Float16* __restrict__ wh = Whh + (size_t)rowA * H + koffA;

    v8f acc = {};
    for (int kk = 0; kk < H; kk += 32) {
        // A = 16x32 weight tile of Wih  (contiguous k = 16h + s, matching B)
        v8h a0 = *(const v8h*)(wi + kk);
        v8h a1 = *(const v8h*)(wi + kk + 8);
        v16h A;
#pragma unroll
        for (int i = 0; i < 8; i++) { A[i] = a0[i]; A[i + 8] = a1[i]; }
        // B = x broadcast across all 16 columns
        v16h Bx = *(const v16h*)(xs + kk + koffB);
        acc = __builtin_amdgcn_wmma_f32_16x16x32_f16(false, A, false, Bx,
                                                     (short)0, acc, false, false);
        // second matrix: Whh @ h accumulated into same C
        v8h c0 = *(const v8h*)(wh + kk);
        v8h c1 = *(const v8h*)(wh + kk + 8);
        v16h A2;
#pragma unroll
        for (int i = 0; i < 8; i++) { A2[i] = c0[i]; A2[i + 8] = c1[i]; }
        v16h Bh = *(const v16h*)(hs + kk + koffB);
        acc = __builtin_amdgcn_wmma_f32_16x16x32_f16(false, A2, false, Bh,
                                                     (short)0, acc, false, false);
    }
    // D: VGPR r -> lanes0-15 hold M=r (N=lane), lanes16-31 hold M=8+r.
    // Column N==0 lives in lane 0 (rows j0..j0+7) and lane 16 (rows j0+8..j0+15).
    if ((lane & 15) == 0) {
        const int mbase = j0 + (lane >> 4) * 8;
#pragma unroll
        for (int r = 0; r < 8; r++) {
            const int j = mbase + r;
            gates[j] = acc[r] + b_ih[j] + b_hh[j];
        }
    }
}

// ---------------- elementwise LSTM cell update ----------------
__global__ void cell_kernel(const float* __restrict__ gates,
                            float* __restrict__ c,
                            _Float16* __restrict__ h16,
                            const int* __restrict__ done) {
    if (done && *done) return;     // frozen after EOS (pre-step flag)
    int i = threadIdx.x;           // 1024 threads
    float ig = gates[i];
    float fg = gates[H + i];
    float gg = gates[2 * H + i];
    float og = gates[3 * H + i];
    float si = 1.f / (1.f + __expf(-ig));
    float sf = 1.f / (1.f + __expf(-fg));
    float so = 1.f / (1.f + __expf(-og));
    float c2 = sf * c[i] + si * tanhf(gg);
    float h2 = so * tanhf(c2);
    c[i] = c2;
    h16[i] = (_Float16)h2;
}

// ---------------- decoder logits via WMMA ----------------
// logits[32000] = out_w @ relu(dec_emb[*tok]) + out_b
// grid = 250 blocks x 8 waves x 16 rows = 32000
__global__ __launch_bounds__(256)
void logits_wmma_kernel(const _Float16* __restrict__ outw,
                        const float* __restrict__ out_b,
                        const float* __restrict__ dec_emb,
                        const int* __restrict__ tok_addr,
                        float* __restrict__ logits) {
    __shared__ __align__(32) _Float16 xs[H];
    const int tid = threadIdx.x;
    const int tok = *tok_addr;
    const float* __restrict__ erow = dec_emb + (size_t)tok * H;
    for (int i = tid; i < H; i += 256) {
        float v = erow[i];
        xs[i] = (_Float16)(v > 0.f ? v : 0.f);
    }
    __syncthreads();

    const int wave = tid >> 5, lane = tid & 31;
    const int j0    = (blockIdx.x * 8 + wave) * 16;
    const int rowA  = j0 + (lane & 15);
    const int koffA = (lane >> 4) * 16;
    const int koffB = (lane >> 4) * 16;
    const _Float16* __restrict__ w = outw + (size_t)rowA * H + koffA;

    v8f acc = {};
    for (int kk = 0; kk < H; kk += 32) {
        v8h a0 = *(const v8h*)(w + kk);
        v8h a1 = *(const v8h*)(w + kk + 8);
        v16h A;
#pragma unroll
        for (int i = 0; i < 8; i++) { A[i] = a0[i]; A[i + 8] = a1[i]; }
        v16h Bx = *(const v16h*)(xs + kk + koffB);
        acc = __builtin_amdgcn_wmma_f32_16x16x32_f16(false, A, false, Bx,
                                                     (short)0, acc, false, false);
    }
    if ((lane & 15) == 0) {
        const int mbase = j0 + (lane >> 4) * 8;
#pragma unroll
        for (int r = 0; r < 8; r++) {
            const int j = mbase + r;
            logits[j] = acc[r] + out_b[j];
        }
    }
}

// ---------------- decoder finalize: argmax + log_softmax + state ----------------
__global__ __launch_bounds__(1024)
void finalize_kernel(const float* __restrict__ logits,
                     float* __restrict__ dout, int step,
                     int* __restrict__ tok, int* __restrict__ done) {
    __shared__ float sv[1024];
    __shared__ int   si[1024];
    __shared__ float sMax, sLse;
    __shared__ int   sArg, sDone0;
    const int tid = threadIdx.x;
    if (tid == 0) sDone0 = *done;

    // pass 1: max + first-index argmax (jnp.argmax semantics)
    float best = -INFINITY; int bidx = 0;
    for (int j = tid; j < OUTV; j += 1024) {
        float v = logits[j];
        if (v > best) { best = v; bidx = j; }
    }
    sv[tid] = best; si[tid] = bidx;
    __syncthreads();
    for (int s = 512; s > 0; s >>= 1) {
        if (tid < s) {
            float vr = sv[tid + s]; int ir = si[tid + s];
            if (vr > sv[tid] || (vr == sv[tid] && ir < si[tid])) {
                sv[tid] = vr; si[tid] = ir;
            }
        }
        __syncthreads();
    }
    if (tid == 0) { sMax = sv[0]; sArg = si[0]; }
    __syncthreads();

    // pass 2: log-sum-exp
    const float m = sMax;
    float a = 0.f;
    for (int j = tid; j < OUTV; j += 1024) a += __expf(logits[j] - m);
    sv[tid] = a;
    __syncthreads();
    for (int s = 512; s > 0; s >>= 1) {
        if (tid < s) sv[tid] += sv[tid + s];
        __syncthreads();
    }
    if (tid == 0) sLse = __logf(sv[0]);
    __syncthreads();

    // pass 3: masked output row + state update
    const int done0 = sDone0;
    const float lse = sLse;
    float* __restrict__ row = dout + 13 + (size_t)step * OUTV;
    for (int j = tid; j < OUTV; j += 1024)
        row[j] = done0 ? 0.f : (logits[j] - m - lse);
    if (tid == 0) {
        const int nxt = sArg;
        dout[1 + step] = done0 ? -1.f : (float)nxt;   // tokens[1+step]
        if (!done0) {
            *tok = nxt;
            if (nxt == 1) *done = 1;                  // EOS
        }
    }
}

// ---------------- host launcher ----------------
extern "C" void kernel_launch(void* const* d_in, const int* in_sizes, int n_in,
                              void* d_out, int out_size, void* d_ws, size_t ws_size,
                              hipStream_t stream) {
    (void)in_sizes; (void)n_in; (void)out_size; (void)ws_size;
    const int*   sentence = (const int*)  d_in[0];
    const float* enc_emb  = (const float*)d_in[1];
    const float* enc_wih  = (const float*)d_in[2];
    const float* enc_whh  = (const float*)d_in[3];
    const float* enc_bih  = (const float*)d_in[4];
    const float* enc_bhh  = (const float*)d_in[5];
    const float* dec_emb  = (const float*)d_in[6];
    const float* dec_wih  = (const float*)d_in[7];
    const float* dec_whh  = (const float*)d_in[8];
    const float* dec_bih  = (const float*)d_in[9];
    const float* dec_bhh  = (const float*)d_in[10];
    const float* out_w    = (const float*)d_in[11];
    const float* out_b    = (const float*)d_in[12];
    float* dout = (float*)d_out;

    char* ws = (char*)d_ws;
    _Float16* enc_wih_h = (_Float16*)(ws + WS_ENC_WIH);
    _Float16* enc_whh_h = (_Float16*)(ws + WS_ENC_WHH);
    _Float16* dec_wih_h = (_Float16*)(ws + WS_DEC_WIH);
    _Float16* dec_whh_h = (_Float16*)(ws + WS_DEC_WHH);
    _Float16* outw_h    = (_Float16*)(ws + WS_OUTW);
    float*    gates     = (float*)   (ws + WS_GATES);
    float*    c_state   = (float*)   (ws + WS_C);
    _Float16* h16       = (_Float16*)(ws + WS_H16);
    float*    logits    = (float*)   (ws + WS_LOGITS);
    int*      tok       = (int*)     (ws + WS_TOK);
    int*      done      = (int*)     (ws + WS_DONE);

    // one-time (per call) f16 weight conversion: halves L2 traffic,
    // makes the whole ~96MB weight set L2-resident on MI455X (192MB L2).
    const int CB = 2048, CT = 256;
    cvt_f16_kernel<<<CB, CT, 0, stream>>>(enc_wih, enc_wih_h, G4 * H);
    cvt_f16_kernel<<<CB, CT, 0, stream>>>(enc_whh, enc_whh_h, G4 * H);
    cvt_f16_kernel<<<CB, CT, 0, stream>>>(dec_wih, dec_wih_h, G4 * H);
    cvt_f16_kernel<<<CB, CT, 0, stream>>>(dec_whh, dec_whh_h, G4 * H);
    cvt_f16_kernel<<<CB, CT, 0, stream>>>(out_w,   outw_h,    OUTV * H);

    init_kernel<<<1, 1024, 0, stream>>>(h16, c_state, tok, done, dout);

    // encoder: 64 sequential LSTM cells
    for (int t = 0; t < SEQ; t++) {
        gates_wmma_kernel<<<32, 256, 0, stream>>>(
            enc_wih_h, enc_whh_h, enc_bih, enc_bhh,
            enc_emb, sentence + t, h16, gates, /*relu=*/0);
        cell_kernel<<<1, 1024, 0, stream>>>(gates, c_state, h16, nullptr);
    }

    // decoder: 12 greedy steps (state frozen after EOS via `done`)
    for (int s = 0; s < DSTEPS; s++) {
        gates_wmma_kernel<<<32, 256, 0, stream>>>(
            dec_wih_h, dec_whh_h, dec_bih, dec_bhh,
            dec_emb, tok, h16, gates, /*relu=*/1);
        cell_kernel<<<1, 1024, 0, stream>>>(gates, c_state, h16, done);
        logits_wmma_kernel<<<250, 256, 0, stream>>>(
            outw_h, out_b, dec_emb, tok, logits);
        finalize_kernel<<<1, 1024, 0, stream>>>(logits, dout, s, tok, done);
    }
}
